// HierarchicalMoEAttention_57690000720290
// MI455X (gfx1250) — matrix operand order repaired and checked by reading, not compiled, not run on hardware
//
#include <hip/hip_runtime.h>
#include <math.h>

// ---------------------------------------------------------------------------
// Types for CDNA5 WMMA (gfx1250, wave32)
// ---------------------------------------------------------------------------
typedef __attribute__((ext_vector_type(16))) __bf16 v16bf;
typedef __attribute__((ext_vector_type(8)))  __bf16 v8bf;
typedef __attribute__((ext_vector_type(8)))  float  v8f;

#define WMMA_BF16(a, b, c) \
  __builtin_amdgcn_wmma_f32_16x16x32_bf16(false, (a), false, (b), (short)0, (c), false, false)

// A-matrix fragment (16x32 bf16): per ISA, lane<16 holds K in {0..7}U{16..23},
// lane>=16 holds K in {8..15}U{24..31}  ->  two 16B chunks at kb and kb+16.
__device__ __forceinline__ v16bf frag_ld_a(const __bf16* rowp, int kb) {
  union { v16bf v; v8bf h[2]; } u;
  u.h[0] = *(const v8bf*)(rowp + kb);
  u.h[1] = *(const v8bf*)(rowp + kb + 16);
  return u.v;
}
// B-matrix fragment (32x16 bf16): same K interleave as A -> two 16B chunks at khalf/2 and khalf/2+16.
__device__ __forceinline__ v16bf frag_ld_b(const __bf16* rowp, int khalf) {
  union { v16bf v; v8bf h[2]; } u;
  u.h[0] = *(const v8bf*)(rowp + khalf / 2);
  u.h[1] = *(const v8bf*)(rowp + khalf / 2 + 16);
  return u.v;
}

// ---------------------------------------------------------------------------
// Unified bf16-WMMA GEMM:  C[M,N] = op(A) * op(W) + bias (+ C if accum)
//   aKM  : 0 -> A is [M,K] row stride lda ; 1 -> A is [K,M] row stride lda
//   wNK  : 1 -> W is [N,K] row stride ldw (the x@w.T case, contiguous K)
//          0 -> W is [K,N] row stride ldw (transposed into LDS on load)
// Tiles: 64x64x32, 128 threads (4 waves in 2x2), wave computes 32x32.
// ---------------------------------------------------------------------------
#define TM 64
#define TN 64
#define TK 32
#define LSTR 48   // padded bf16 row stride (96B -> 16B/32B aligned chunks)

__global__ void __launch_bounds__(128)
gemm_wmma(const float* __restrict__ A, int lda, int aKM,
          const float* __restrict__ W, int ldw, int wNK,
          const float* __restrict__ bias,
          float* __restrict__ C, int ldc,
          int M, int N, int K, int accum)
{
  __shared__ __bf16 As[TM][LSTR];
  __shared__ __bf16 Ws[TN][LSTR];

  const int t    = threadIdx.x;
  const int lane = t & 31;
  const int wid  = t >> 5;
  const int wm   = (wid & 1) * 32;
  const int wn   = (wid >> 1) * 32;
  const int lr   = lane & 15;
  const int kb   = (lane >> 4) * 8;     // A interleave base
  const int kh   = (lane >> 4) * 16;    // B half offset; frag_ld_b halves it to the interleave base
  const int rowHi = (lane >> 4) * 8;    // C layout row offset

  const int mBase = blockIdx.x * TM;
  const int nBase = blockIdx.y * TN;

  v8f acc[2][2] = {};

  for (int kBase = 0; kBase < K; kBase += TK) {
    if (kBase + TK < K) {
      const float* pf = aKM ? &A[(size_t)(kBase + TK) * lda + mBase]
                            : &A[(size_t)mBase * lda + kBase + TK];
      __builtin_prefetch(pf, 0, 1);
    }
    // stage A tile (f32 -> bf16)
    for (int it = 0; it < 16; ++it) {
      int e = t + it * 128;
      int r = e >> 5, c = e & 31;
      float v = aKM ? A[(size_t)(kBase + c) * lda + (mBase + r)]
                    : A[(size_t)(mBase + r) * lda + (kBase + c)];
      As[r][c] = (__bf16)v;
    }
    // stage W tile as Ws[n][k] (f32 -> bf16), N-guarded
    for (int it = 0; it < 16; ++it) {
      int e = t + it * 128;
      int r = e >> 5, c = e & 31;
      int n = nBase + r;
      float v = 0.f;
      if (n < N)
        v = wNK ? W[(size_t)n * ldw + (kBase + c)]
                : W[(size_t)(kBase + c) * ldw + n];
      Ws[r][c] = (__bf16)v;
    }
    __syncthreads();

    v16bf a0 = frag_ld_a(&As[wm + lr][0], kb);
    v16bf a1 = frag_ld_a(&As[wm + 16 + lr][0], kb);
    v16bf b0 = frag_ld_b(&Ws[wn + lr][0], kh);
    v16bf b1 = frag_ld_b(&Ws[wn + 16 + lr][0], kh);
    acc[0][0] = WMMA_BF16(a0, b0, acc[0][0]);
    acc[0][1] = WMMA_BF16(a0, b1, acc[0][1]);
    acc[1][0] = WMMA_BF16(a1, b0, acc[1][0]);
    acc[1][1] = WMMA_BF16(a1, b1, acc[1][1]);
    __syncthreads();
  }

  for (int i = 0; i < 2; ++i)
    for (int j = 0; j < 2; ++j)
      for (int r = 0; r < 8; ++r) {
        int row = mBase + wm + i * 16 + r + rowHi;
        int col = nBase + wn + j * 16 + lr;
        if (col < N) {
          float v = acc[i][j][r];
          if (bias)  v += bias[col];
          if (accum) v += C[(size_t)row * ldc + col];
          C[(size_t)row * ldc + col] = v;
        }
      }
}

// ---------------------------------------------------------------------------
// Flash-style attention with WMMA. q/k/v/out are [B*S, E] token-major; a head
// is a [S, HD] slice with row stride E. Optional multiplicative mask[B,S]
// applied to the (already scaled) scores before softmax (sparse expert).
// Block: 128 threads (4 waves), 64 query rows; loops over 64-key tiles with
// online softmax; V is transposed into LDS for contiguous B fragments.
// ---------------------------------------------------------------------------
__global__ void __launch_bounds__(128)
flash_attn_kernel(const float* __restrict__ Q, const float* __restrict__ Kp,
                  const float* __restrict__ V, const float* __restrict__ mask,
                  float* __restrict__ O, float scale,
                  int S_, int E_, int H_, int HD_)
{
  __shared__ __bf16 Qs[64][80];
  __shared__ __bf16 Ks[64][80];
  __shared__ __bf16 Vts[64][80];
  __shared__ float  Sf[64][66];
  __shared__ __bf16 Ps[64][80];
  __shared__ float  alphaS[64];
  __shared__ float  lS[64];

  const int t    = threadIdx.x;
  const int lane = t & 31;
  const int w    = t >> 5;
  const int lr   = lane & 15;
  const int kb   = (lane >> 4) * 8;
  const int kh   = (lane >> 4) * 16;
  const int rowHi = (lane >> 4) * 8;

  const int qBase = blockIdx.x * 64;
  const int b = blockIdx.y / H_;
  const int h = blockIdx.y % H_;
  const size_t base = (size_t)b * S_ * E_ + (size_t)h * HD_;
  const float* qp = Q + base;
  const float* kp = Kp + base;
  const float* vp = V + base;
  float* op = O + base;

  for (int it = 0; it < 32; ++it) {            // 64x64 Q tile, pre-scaled
    int e = t + it * 128;
    int r = e >> 6, d = e & 63;
    Qs[r][d] = (__bf16)(qp[(size_t)(qBase + r) * E_ + d] * scale);
  }

  float m_i = -1e30f, l_i = 0.f;               // owned by thread t (t < 64)
  v8f acc_o[4] = {};

  const int nKT = S_ / 64;
  for (int kt = 0; kt < nKT; ++kt) {
    __syncthreads();
    for (int it = 0; it < 32; ++it) {
      int e = t + it * 128;
      int r = e >> 6, d = e & 63;
      Ks[r][d]  = (__bf16)kp[(size_t)(kt * 64 + r) * E_ + d];
      Vts[d][r] = (__bf16)vp[(size_t)(kt * 64 + r) * E_ + d];
    }
    __syncthreads();

    // scores: this wave's 16 query rows x 64 keys
    v8f s[4] = {};
    for (int ks = 0; ks < 64; ks += 32) {
      v16bf a = frag_ld_a(&Qs[w * 16 + lr][ks], kb);
      for (int n = 0; n < 4; ++n) {
        v16bf bf = frag_ld_b(&Ks[n * 16 + lr][ks], kh);
        s[n] = WMMA_BF16(a, bf, s[n]);
      }
    }
    for (int n = 0; n < 4; ++n) {
      float mv = 1.f;
      if (mask) mv = mask[(size_t)b * S_ + kt * 64 + n * 16 + lr];
      for (int r = 0; r < 8; ++r)
        Sf[w * 16 + r + rowHi][n * 16 + lr] = s[n][r] * mv;
    }
    __syncthreads();

    if (t < 64) {                               // per-row online softmax
      float mt = -1e30f;
      for (int c = 0; c < 64; ++c) mt = fmaxf(mt, Sf[t][c]);
      float mn = fmaxf(m_i, mt);
      float al = expf(m_i - mn);
      float ls = 0.f;
      for (int c = 0; c < 64; ++c) {
        float p = expf(Sf[t][c] - mn);
        Ps[t][c] = (__bf16)p;
        ls += p;
      }
      l_i = l_i * al + ls;
      m_i = mn;
      alphaS[t] = al;
    }
    __syncthreads();

    for (int n = 0; n < 4; ++n)                 // rescale running output
      for (int r = 0; r < 8; ++r)
        acc_o[n][r] *= alphaS[w * 16 + r + rowHi];

    for (int ks = 0; ks < 64; ks += 32) {       // P x V
      v16bf a = frag_ld_a(&Ps[w * 16 + lr][ks], kb);
      for (int n = 0; n < 4; ++n) {
        v16bf bf = frag_ld_b(&Vts[n * 16 + lr][ks], kh);
        acc_o[n] = WMMA_BF16(a, bf, acc_o[n]);
      }
    }
  }

  if (t < 64) lS[t] = l_i;
  __syncthreads();
  for (int n = 0; n < 4; ++n)
    for (int r = 0; r < 8; ++r) {
      int row = w * 16 + r + rowHi;
      op[(size_t)(qBase + row) * E_ + (n * 16 + lr)] = acc_o[n][r] / lS[row];
    }
}

// ---------------------------------------------------------------------------
// Elementwise / reduction helper kernels
// ---------------------------------------------------------------------------
__global__ void ln_kernel(const float* __restrict__ in, float* __restrict__ out,
                          const float* __restrict__ g, const float* __restrict__ be,
                          int D)
{
  __shared__ float red[256];
  const int row = blockIdx.x;
  const int t = threadIdx.x;
  const float* x = in + (size_t)row * D;
  float s = 0.f;
  for (int c = t; c < D; c += 256) s += x[c];
  red[t] = s; __syncthreads();
  for (int o = 128; o > 0; o >>= 1) { if (t < o) red[t] += red[t + o]; __syncthreads(); }
  float mean = red[0] / (float)D;
  __syncthreads();
  float v = 0.f;
  for (int c = t; c < D; c += 256) { float d = x[c] - mean; v += d * d; }
  red[t] = v; __syncthreads();
  for (int o = 128; o > 0; o >>= 1) { if (t < o) red[t] += red[t + o]; __syncthreads(); }
  float inv = rsqrtf(red[0] / (float)D + 1e-5f);
  for (int c = t; c < D; c += 256)
    out[(size_t)row * D + c] = (x[c] - mean) * inv * g[c] + be[c];
}

__global__ void relu_kernel(float* p, size_t n) {
  size_t i = (size_t)blockIdx.x * blockDim.x + threadIdx.x;
  size_t st = (size_t)gridDim.x * blockDim.x;
  for (; i < n; i += st) p[i] = fmaxf(p[i], 0.f);
}

__global__ void gelu_kernel(float* p, size_t n) {   // exact erf GELU
  size_t i = (size_t)blockIdx.x * blockDim.x + threadIdx.x;
  size_t st = (size_t)gridDim.x * blockDim.x;
  for (; i < n; i += st) {
    float x = p[i];
    p[i] = 0.5f * x * (1.f + erff(x * 0.70710678118654752f));
  }
}

__global__ void elu1_kernel(float* p, size_t n) {   // elu(x)+1
  size_t i = (size_t)blockIdx.x * blockDim.x + threadIdx.x;
  size_t st = (size_t)gridDim.x * blockDim.x;
  for (; i < n; i += st) {
    float x = p[i];
    p[i] = (x > 0.f) ? (x + 1.f) : expf(x);
  }
}

__global__ void softmax_rows_kernel(float* p, int rows, int cols) {
  int i = blockIdx.x * blockDim.x + threadIdx.x;
  int st = gridDim.x * blockDim.x;
  for (; i < rows; i += st) {
    float* r = p + (size_t)i * cols;
    float m = r[0];
    for (int c = 1; c < cols; ++c) m = fmaxf(m, r[c]);
    float s = 0.f;
    for (int c = 0; c < cols; ++c) { r[c] = expf(r[c] - m); s += r[c]; }
    float inv = 1.f / s;
    for (int c = 0; c < cols; ++c) r[c] *= inv;
  }
}

__global__ void scaled_accum_kernel(float* __restrict__ dst, const float* __restrict__ src,
                                    const float* __restrict__ prob, int pstride, int j,
                                    int E_, size_t n)
{
  size_t i = (size_t)blockIdx.x * blockDim.x + threadIdx.x;
  size_t st = (size_t)gridDim.x * blockDim.x;
  for (; i < n; i += st)
    dst[i] += prob[(i / E_) * pstride + j] * src[i];
}

__global__ void topk_mask_kernel(const float* __restrict__ imp, float* __restrict__ mask,
                                 int S_, int k)
{
  __shared__ float vals[1024];
  __shared__ float rv[256];
  __shared__ int   ri[256];
  const int b = blockIdx.x, t = threadIdx.x;
  for (int s = t; s < S_; s += 256) { vals[s] = imp[(size_t)b * S_ + s]; mask[(size_t)b * S_ + s] = 0.f; }
  __syncthreads();
  for (int it = 0; it < k; ++it) {
    float best = -1e30f; int bi = 0x7fffffff;
    for (int s = t; s < S_; s += 256)
      if (vals[s] > best || (vals[s] == best && s < bi)) { best = vals[s]; bi = s; }
    rv[t] = best; ri[t] = bi; __syncthreads();
    for (int o = 128; o > 0; o >>= 1) {
      if (t < o)
        if (rv[t + o] > rv[t] || (rv[t + o] == rv[t] && ri[t + o] < ri[t])) { rv[t] = rv[t + o]; ri[t] = ri[t + o]; }
      __syncthreads();
    }
    if (t == 0) { mask[(size_t)b * S_ + ri[0]] = 1.f; vals[ri[0]] = -3e30f; }
    __syncthreads();
  }
}

__global__ void dwconv3_kernel(const float* __restrict__ x, const float* __restrict__ dw,
                               const float* __restrict__ db, float* __restrict__ out,
                               int S_, int E_, size_t n)
{
  size_t i = (size_t)blockIdx.x * blockDim.x + threadIdx.x;
  size_t st = (size_t)gridDim.x * blockDim.x;
  for (; i < n; i += st) {
    int c = (int)(i % E_);
    size_t nr = i / E_;
    int b = (int)(nr / S_), s = (int)(nr % S_);
    float acc = db[c];
    for (int tt = -1; tt <= 1; ++tt) {
      int ss = s + tt;
      if (ss >= 0 && ss < S_)
        acc += x[((size_t)b * S_ + ss) * E_ + c] * dw[c * 3 + tt + 1];
    }
    out[i] = acc;
  }
}

__global__ void ksum_kernel(const float* __restrict__ kp, float* __restrict__ ks,
                            int KD_, int S_)
{
  int m = blockIdx.x * blockDim.x + threadIdx.x;
  if (m < KD_) {
    float s = 0.f;
    for (int l = 0; l < S_; ++l) s += kp[(size_t)l * KD_ + m];
    ks[m] = s;
  }
}

__global__ void perf_zdiv_kernel(const float* __restrict__ qp, const float* __restrict__ ks,
                                 float* __restrict__ y, int KD_, int HD_, int ldy)
{
  __shared__ float red[256];
  const int l = blockIdx.x, t = threadIdx.x;
  float s = 0.f;
  for (int m = t; m < KD_; m += 256) s += qp[(size_t)l * KD_ + m] * ks[m];
  red[t] = s; __syncthreads();
  for (int o = 128; o > 0; o >>= 1) { if (t < o) red[t] += red[t + o]; __syncthreads(); }
  float inv = 1.f / (red[0] + 1e-8f);
  if (t < HD_) y[(size_t)l * ldy + t] *= inv;
}

// ---------------------------------------------------------------------------
// Orchestration
// ---------------------------------------------------------------------------
extern "C" void kernel_launch(void* const* d_in, const int* in_sizes, int n_in,
                              void* d_out, int out_size, void* d_ws, size_t ws_size,
                              hipStream_t stream)
{
  (void)in_sizes; (void)n_in; (void)out_size; (void)ws_size;
  constexpr int Bc = 4, Sc = 1024, Ec = 1024, Hc = 16, HDc = 64, KDc = 256;
  constexpr int NEc = 4, NSc = 2, TOPKc = 10;
  const int Nn = Bc * Sc;                      // 4096 tokens
  const size_t NE_ = (size_t)Nn * Ec;          // [N,E] elements

  const float* X = (const float*)d_in[0];
  float* out = (float*)d_out;
  float* wsf = (float*)d_ws;
  float* R[8];
  for (int i = 0; i < 8; ++i) R[i] = wsf + (size_t)i * NE_;
  float* probs1 = wsf + 8 * NE_;
  float* probs2 = probs1 + (size_t)Nn * NEc;
  float* impb   = probs2 + (size_t)Nn * NSc;
  float* maskb  = impb + Nn;
  float* qpb    = maskb + (size_t)Bc * Sc;
  float* kpb    = qpb + (size_t)Sc * KDc;
  float* kvb    = kpb + (size_t)Sc * KDc;
  float* ksumb  = kvb + (size_t)KDc * HDc;

  auto in = [&](int i) { return (const float*)d_in[i]; };
  auto G = [&](const float* A, int lda, int aKM, const float* W, int ldw, int wNK,
               const float* bias, float* C, int ldc, int M, int N, int K, int accum) {
    dim3 grid(M / 64, (N + 63) / 64);
    gemm_wmma<<<grid, 128, 0, stream>>>(A, lda, aKM, W, ldw, wNK, bias, C, ldc, M, N, K, accum);
  };
  const int eb[4] = {1, 11, 23, 31};           // expert param base indices

  hipMemsetAsync(out, 0, NE_ * sizeof(float), stream);

  // ---- level-1 router: softmax(relu(ln(x@w1.T+b1))@w2.T+b2) over NE ----
  G(X, Ec, 0, in(93), Ec, 1, in(94), R[5], 512, Nn, 512, Ec, 0);
  ln_kernel<<<Nn, 256, 0, stream>>>(R[5], R[5], in(95), in(96), 512);
  relu_kernel<<<8192, 256, 0, stream>>>(R[5], (size_t)Nn * 512);
  G(R[5], 512, 0, in(97), 512, 1, in(98), probs1, NEc, Nn, NEc, 512, 0);
  softmax_rows_kernel<<<(Nn + 255) / 256, 256, 0, stream>>>(probs1, Nn, NEc);

  for (int ei = 0; ei < NEc; ++ei) {
    const int e0 = eb[ei];
    const float *qw = in(e0 + 0), *qb = in(e0 + 1), *kw = in(e0 + 2), *kb_ = in(e0 + 3);
    const float *vw = in(e0 + 4), *vb = in(e0 + 5), *ow = in(e0 + 6), *ob = in(e0 + 7);

    // Q,K,V projections
    G(X, Ec, 0, qw, Ec, 1, qb, R[0], Ec, Nn, Ec, Ec, 0);
    G(X, Ec, 0, kw, Ec, 1, kb_, R[1], Ec, Nn, Ec, Ec, 0);
    G(X, Ec, 0, vw, Ec, 1, vb, R[2], Ec, Nn, Ec, Ec, 0);

    const float* maskPtr = nullptr;
    if (ei == 0) {                                   // sparse: top-k mask
      G(X, Ec, 0, in(9), Ec, 1, in(10), impb, 1, Nn, 1, Ec, 0);
      topk_mask_kernel<<<Bc, 256, 0, stream>>>(impb, maskb, Sc, TOPKc);
      maskPtr = maskb;
    }

    if (ei == 1) {                                   // performer (linear attention)
      for (int b = 0; b < Bc; ++b)
        for (int h = 0; h < Hc; ++h) {
          size_t off = (size_t)b * Sc * Ec + (size_t)h * HDc;
          const float *qbh = R[0] + off, *kbh = R[1] + off, *vbh = R[2] + off;
          float* obh = R[3] + off;
          G(qbh, Ec, 0, in(19), HDc, 1, in(20), qpb, KDc, Sc, KDc, HDc, 0);
          elu1_kernel<<<1024, 256, 0, stream>>>(qpb, (size_t)Sc * KDc);
          G(kbh, Ec, 0, in(21), HDc, 1, in(22), kpb, KDc, Sc, KDc, HDc, 0);
          elu1_kernel<<<1024, 256, 0, stream>>>(kpb, (size_t)Sc * KDc);
          ksum_kernel<<<1, 256, 0, stream>>>(kpb, ksumb, KDc, Sc);
          G(kpb, KDc, 1, vbh, Ec, 0, nullptr, kvb, HDc, KDc, HDc, Sc, 0);   // kv = kp^T v
          G(qpb, KDc, 0, kvb, HDc, 0, nullptr, obh, Ec, Sc, HDc, KDc, 0);   // qp kv
          perf_zdiv_kernel<<<Sc, 256, 0, stream>>>(qpb, ksumb, obh, KDc, HDc, Ec);
        }
    } else {                                         // softmax attention (flash)
      flash_attn_kernel<<<dim3(Sc / 64, Bc * Hc), 128, 0, stream>>>(
          R[0], R[1], R[2], maskPtr, R[3], 0.125f, Sc, Ec, Hc, HDc);
    }

    // output projection
    G(R[3], Ec, 0, ow, Ec, 1, ob, R[4], Ec, Nn, Ec, Ec, 0);
    float* eo = R[4];

    if (ei == 3) {                                   // temporal: conv branch + fuse
      dwconv3_kernel<<<16384, 256, 0, stream>>>(X, in(39), in(40), R[0], Sc, Ec, NE_);
      G(R[0], Ec, 0, in(41), Ec, 1, in(42), R[1], Ec, Nn, Ec, Ec, 0);
      gelu_kernel<<<16384, 256, 0, stream>>>(R[1], NE_);
      G(R[4], Ec, 0, in(43),       2 * Ec, 1, in(44), R[5], Ec, Nn, Ec, Ec, 0); // fw[:, :E]
      G(R[1], Ec, 0, in(43) + Ec,  2 * Ec, 1, nullptr, R[5], Ec, Nn, Ec, Ec, 1); // fw[:, E:]
      eo = R[5];
    }

    // ---- level-2 router ----
    const int rb = 99 + ei * 6;
    G(X, Ec, 0, in(rb + 0), Ec, 1, in(rb + 1), R[0], 256, Nn, 256, Ec, 0);
    ln_kernel<<<Nn, 256, 0, stream>>>(R[0], R[0], in(rb + 2), in(rb + 3), 256);
    relu_kernel<<<4096, 256, 0, stream>>>(R[0], (size_t)Nn * 256);
    G(R[0], 256, 0, in(rb + 4), 256, 1, in(rb + 5), probs2, NSc, Nn, NSc, 256, 0);
    softmax_rows_kernel<<<(Nn + 255) / 256, 256, 0, stream>>>(probs2, Nn, NSc);

    // ---- sub-experts, weighted combine ----
    hipMemsetAsync(R[2], 0, NE_ * sizeof(float), stream);
    for (int j = 0; j < NSc; ++j) {
      const int sb = 45 + (ei * NSc + j) * 6;
      G(eo, Ec, 0, in(sb + 0), Ec, 1, in(sb + 1), R[6], Ec, Nn, Ec, Ec, 0);
      ln_kernel<<<Nn, 256, 0, stream>>>(R[6], R[6], in(sb + 2), in(sb + 3), Ec);
      gelu_kernel<<<16384, 256, 0, stream>>>(R[6], NE_);
      G(R[6], Ec, 0, in(sb + 4), Ec, 1, in(sb + 5), R[7], Ec, Nn, Ec, Ec, 0);
      scaled_accum_kernel<<<16384, 256, 0, stream>>>(R[2], R[7], probs2, NSc, j, Ec, NE_);
    }
    scaled_accum_kernel<<<16384, 256, 0, stream>>>(out, R[2], probs1, NEc, ei, Ec, NE_);
  }
}
